// GGPLNumericEmbedding_45165876085582
// MI455X (gfx1250) — matrix operand order repaired and checked by reading, not compiled, not run on hardware
//
#include <hip/hip_runtime.h>
#include <hip/hip_bf16.h>
#include <math.h>

// ---------------- problem constants ----------------
#define B_    8192
#define D_    256
#define K_    63
#define F_    64            // K+1 basis funcs per feature
#define E_    64            // emb dim per feature
#define DE_   16384         // D_*E_
#define PROJ_SCALE 0.1f

// d_out layout (floats): [x_emb | embeddings | breakpoints]
static const size_t XEMB_OFF = 0;
static const size_t EMB_OFF  = (size_t)B_ * D_;                    // 2,097,152
static const size_t BP_OFF   = EMB_OFF + (size_t)B_ * D_ * E_;     // 136,314,880

// LDS tile row stride (bf16 units): 144 B = 16B-aligned rows for async B128,
// 36 words/row spreads banks (9*m mod 16 is a permutation).
#define PAD_ 72

// dynamic LDS layout (bytes)
#define LS_BASIS_OFF 0
#define LS_EMB_OFF   (64 * PAD_ * 2)                    //  9216
#define LS_W_OFF     (2 * 64 * PAD_ * 2)                // 18432 (2 buffers of 9216)
#define LS_WBUF_SZ   (64 * PAD_ * 2)
#define LS_P_OFF     (4 * 64 * PAD_ * 2)                // 36864 (2 buffers of 36864)
#define LS_PBUF_SZ   (256 * PAD_ * 2)
#define LS_TOTAL     (LS_P_OFF + 2 * LS_PBUF_SZ)        // 110592

typedef __bf16 v16bf __attribute__((ext_vector_type(16)));
typedef float  v8f   __attribute__((ext_vector_type(8)));

struct Frag { union { unsigned int u[8]; v16bf v; }; };

// ---- A-matrix fragment (16x32 bf16, M x K) from LDS, row-major, stride PAD_ ----
// ISA 7.12.2: lanes 0-15 = M, V0..3 hold K 0..7 (pairs), V4..7 hold K 16..23;
// lanes 16-31 hold K 8..15 / 24..31.
__device__ __forceinline__ void loadA(Frag& f, const __bf16* base, int mrow0, int kb, int lane) {
  const int m  = lane & 15;
  const int hq = lane >> 4;
  const __bf16* rowp = base + (mrow0 + m) * PAD_;
  const int k0 = kb + 8 * hq;
#pragma unroll
  for (int v = 0; v < 4; ++v)
    f.u[v] = *(const unsigned int*)(rowp + k0 + 2 * v);
#pragma unroll
  for (int v = 0; v < 4; ++v)
    f.u[4 + v] = *(const unsigned int*)(rowp + k0 + 16 + 2 * v);
}

// ---- B-matrix fragment (32x16 bf16, K x N) from LDS stored [n][k] (K contiguous) ----
// lane = N; lanes 0-15 hold K kb..kb+7 / kb+16..kb+23 across V0..7 (pairs), lanes 16-31 hold K kb+8..kb+15 / kb+24..kb+31.
__device__ __forceinline__ void loadB(Frag& f, const __bf16* base, int nrow0, int kb, int lane) {
  const int nn = lane & 15;
  const int hq = lane >> 4;
  const __bf16* rowp = base + (nrow0 + nn) * PAD_;
  const int k0 = kb + 8 * hq;
#pragma unroll
  for (int v = 0; v < 8; ++v)
    f.u[v] = *(const unsigned int*)(rowp + k0 + 2 * v + (v >= 4 ? 8 : 0));
}

__device__ __forceinline__ v8f wmma_bf16(const Frag& a, const Frag& b, v8f c) {
  return __builtin_amdgcn_wmma_f32_16x16x32_bf16(
      /*neg_a=*/false, a.v, /*neg_b=*/false, b.v,
      /*c_mod=*/(short)0, c, /*reuse_a=*/false, /*reuse_b=*/false);
}

// ---- async global -> LDS copy, 16 bytes per lane, tracked by ASYNCcnt ----
// LDS dest offset = low 32 bits of the generic pointer (LDS aperture maps addr[31:0]).
__device__ __forceinline__ void async_copy_b128(void* lds_elem, const void* g) {
  unsigned loff = (unsigned)(uintptr_t)lds_elem;
  asm volatile("global_load_async_to_lds_b128 %0, %1, off"
               :: "v"(loff), "v"(g) : "memory");
}
__device__ __forceinline__ void wait_async_le5() {
  asm volatile("s_wait_asynccnt 0x5" ::: "memory");
}
__device__ __forceinline__ void wait_async_0() {
  asm volatile("s_wait_asynccnt 0x0" ::: "memory");
}

// ---------------- kernel 1: breakpoints (softplus cumsum) ----------------
__global__ void ggpl_bp_kernel(const float* __restrict__ bpmin,
                               const float* __restrict__ draw,
                               float* __restrict__ bp_out) {
  int d = blockIdx.x * blockDim.x + threadIdx.x;
  if (d >= D_) return;
  float base = bpmin[d];
  float c = 0.f;
  for (int k = 0; k < K_; ++k) {
    float x = draw[d * K_ + k];
    float sp = (x > 15.f) ? x : log1pf(expf(x));   // stable softplus
    c += sp;
    bp_out[d * K_ + k] = base + c;
  }
}

// ---- kernel 2: one-time f32 -> bf16 conversion; basis_weight also transposed ----
// projbf[o][i]   = bf16(proj_w[o][i])               (layout unchanged, [256][DE])
// wbf_t[d][e][f] = bf16(basis_weight[d][f][e])      (transposed so K=f is contiguous)
__global__ void ggpl_cvt_kernel(const float* __restrict__ projw,
                                const float* __restrict__ bw,
                                __bf16* __restrict__ projbf,
                                __bf16* __restrict__ wbf_t) {
  const int n1 = D_ * DE_;        // 4,194,304
  const int n2 = D_ * F_ * E_;    // 1,048,576
  for (int i = blockIdx.x * blockDim.x + threadIdx.x; i < n1 + n2;
       i += gridDim.x * blockDim.x) {
    if (i < n1) {
      projbf[i] = (__bf16)projw[i];
    } else {
      int j = i - n1;
      int d = j >> 12;            // /4096
      int r = j & 4095;
      int f = r >> 6, e = r & 63;
      wbf_t[((size_t)d * E_ + e) * F_ + f] = (__bf16)bw[j];
    }
  }
}

// ---------------- kernel 3: fused basis -> emb GEMM -> proj GEMM ----------------
// 128 workgroups x 512 threads (16 wave32). Workgroup owns 64 batch rows.
// wave w: row-strip rs = w&3, N-group ng = w>>2 (also e-tile of the emb GEMM).
// ls_p / ls_w are double-buffered and filled by async global->LDS DMA one
// iteration ahead (5 b128 issues/thread), overlapped with WMMA work.
__launch_bounds__(512, 1)
__global__ void ggpl_fused_kernel(const float* __restrict__ x_num,
                                  const float* __restrict__ basis_bias,
                                  const float* __restrict__ proj_b,
                                  const float* __restrict__ bp,       // [D,K]
                                  const __bf16* __restrict__ wbf_t,   // [D,E,F]
                                  const __bf16* __restrict__ projbf,  // [256, DE]
                                  float* __restrict__ xemb_out,
                                  float* __restrict__ emb_out) {
  extern __shared__ __align__(16) unsigned char smem[];
  __bf16* ls_basis = (__bf16*)(smem + LS_BASIS_OFF);        // [64][PAD_]  (row, f)
  __bf16* ls_emb   = (__bf16*)(smem + LS_EMB_OFF);          // [64][PAD_]  (row, e)
  // NOTE: buffer pointers computed at runtime (no LDS-derived aggregate
  // initializers -- those become unrelocatable addrspacecast static inits).

  const int t    = threadIdx.x;
  const int lane = t & 31;
  const int wave = t >> 5;
  const int rs   = wave & 3;
  const int ng   = wave >> 2;
  const int n    = lane & 15;
  const int hi   = lane >> 4;
  const int r0   = blockIdx.x * 64;

  v8f acc[4];
#pragma unroll
  for (int i = 0; i < 4; ++i)
#pragma unroll
    for (int r = 0; r < 8; ++r) acc[i][r] = 0.f;

  // -- issue async staging of slice d into buffer buf: 4 b128 (proj) + 1 b128 (W) --
  auto issue_stage = [&](int d, int buf) {
    __bf16* lsp = (__bf16*)(smem + LS_P_OFF + buf * LS_PBUF_SZ);  // [o][PAD_]
    __bf16* lsw = (__bf16*)(smem + LS_W_OFF + buf * LS_WBUF_SZ);  // [e][PAD_]
    // proj slice: 256 rows x 64 bf16 -> [o][e], 8 chunks of 16B per row
#pragma unroll
    for (int i = 0; i < 4; ++i) {
      int idx = t + i * 512;
      int o = idx >> 3, c = idx & 7;
      async_copy_b128(lsp + o * PAD_ + c * 8,
                      projbf + (size_t)o * DE_ + (size_t)d * E_ + c * 8);
    }
    // W_d transposed slice: 64 rows (e) x 64 bf16 (f)
    {
      int e = t >> 3, c = t & 7;
      async_copy_b128(lsw + e * PAD_ + c * 8,
                      wbf_t + ((size_t)d * E_ + e) * F_ + c * 8);
    }
  };

  issue_stage(0, 0);

  for (int d = 0; d < D_; ++d) {
    const int buf = d & 1;
    __bf16* lsp = (__bf16*)(smem + LS_P_OFF + buf * LS_PBUF_SZ);
    __bf16* lsw = (__bf16*)(smem + LS_W_OFF + buf * LS_WBUF_SZ);

    if (d + 1 < D_) issue_stage(d + 1, (d + 1) & 1);   // pipeline next slice

    // ---- stage basis tile synchronously (computed, can't DMA) ----
#pragma unroll
    for (int i = 0; i < 8; ++i) {
      int idx = t + i * 512;
      int row = idx >> 6, f = idx & 63;
      float x = x_num[(size_t)(r0 + row) * D_ + d];
      float v = (f == 0) ? x : fmaxf(x - bp[d * K_ + (f - 1)], 0.f);
      ls_basis[row * PAD_ + f] = (__bf16)v;
    }

    // wait for THIS iteration's async transfers (5 newer ones may stay in flight)
    if (d + 1 < D_) wait_async_le5(); else wait_async_0();
    __syncthreads();

    // ---- emb GEMM: tile (rs, e-tile ng): C16x16 = basis(16x64) x W(64x16) ----
    Frag a0, a1, b0, b1;
    loadA(a0, ls_basis, rs * 16, 0, lane);
    loadA(a1, ls_basis, rs * 16, 32, lane);
    loadB(b0, lsw, ng * 16, 0, lane);
    loadB(b1, lsw, ng * 16, 32, lane);
    v8f emb;
#pragma unroll
    for (int r = 0; r < 8; ++r) emb[r] = 0.f;
    emb = wmma_bf16(a0, b0, emb);
    emb = wmma_bf16(a1, b1, emb);

    float bias = basis_bias[d * E_ + ng * 16 + n];
#pragma unroll
    for (int r = 0; r < 8; ++r) {
      float v = emb[r] + bias;
      int m = r + 8 * hi;
      int row = rs * 16 + m;
      emb_out[((size_t)(r0 + row) * D_ + d) * E_ + ng * 16 + n] = v;   // stream f32 out
      ls_emb[row * PAD_ + ng * 16 + n] = (__bf16)v;                    // keep bf16 for proj
    }
    __syncthreads();

    // ---- proj accumulation: acc[64x256] += emb(64x64) x P_d(64x256) ----
    Frag ea0, ea1;
    loadA(ea0, ls_emb, rs * 16, 0, lane);
    loadA(ea1, ls_emb, rs * 16, 32, lane);
#pragma unroll
    for (int nt = 0; nt < 4; ++nt) {
      Frag p0, p1;
      loadB(p0, lsp, (ng * 4 + nt) * 16, 0, lane);
      loadB(p1, lsp, (ng * 4 + nt) * 16, 32, lane);
      acc[nt] = wmma_bf16(ea0, p0, acc[nt]);
      acc[nt] = wmma_bf16(ea1, p1, acc[nt]);
    }
    __syncthreads();
  }

  // ---- epilogue: x_emb = x + 0.1*(acc + proj_b) ----
#pragma unroll
  for (int nt = 0; nt < 4; ++nt) {
    int o = (ng * 4 + nt) * 16 + n;
    float pb = proj_b[o];
#pragma unroll
    for (int r = 0; r < 8; ++r) {
      int m = r + 8 * hi;
      int row = rs * 16 + m;
      size_t gi = (size_t)(r0 + row) * D_ + o;
      xemb_out[gi] = x_num[gi] + PROJ_SCALE * (acc[nt][r] + pb);
    }
  }
}

// ---------------- host launcher ----------------
extern "C" void kernel_launch(void* const* d_in, const int* in_sizes, int n_in,
                              void* d_out, int out_size, void* d_ws, size_t ws_size,
                              hipStream_t stream) {
  (void)in_sizes; (void)n_in; (void)out_size; (void)ws_size;
  const float* x_num        = (const float*)d_in[0];
  const float* basis_weight = (const float*)d_in[1];
  const float* basis_bias   = (const float*)d_in[2];
  const float* proj_w       = (const float*)d_in[3];
  const float* proj_b       = (const float*)d_in[4];
  const float* bpmin        = (const float*)d_in[5];
  const float* bpraw        = (const float*)d_in[6];

  float* out    = (float*)d_out;
  float* xemb   = out + XEMB_OFF;
  float* embo   = out + EMB_OFF;
  float* bpout  = out + BP_OFF;

  // workspace: bf16 proj_w copy (8 MB) + bf16 transposed basis_weight (2 MB)
  __bf16* projbf = (__bf16*)d_ws;
  __bf16* wbf_t  = projbf + (size_t)D_ * DE_;

  ggpl_bp_kernel<<<1, 256, 0, stream>>>(bpmin, bpraw, bpout);
  ggpl_cvt_kernel<<<1024, 256, 0, stream>>>(proj_w, basis_weight, projbf, wbf_t);
  ggpl_fused_kernel<<<B_ / 64, 512, LS_TOTAL, stream>>>(x_num, basis_bias, proj_b,
                                                        bpout, wbf_t, projbf,
                                                        xemb, embo);
}
